// GATLayer_67723044323855
// MI455X (gfx1250) — matrix operand order repaired and checked by reading, not compiled, not run on hardware
//
#include <hip/hip_runtime.h>

typedef __attribute__((ext_vector_type(16))) _Float16     v16h;
typedef __attribute__((ext_vector_type(8)))  float        v8f;
typedef __attribute__((ext_vector_type(4)))  float        v4f;
typedef __attribute__((ext_vector_type(4)))  int          v4i;
typedef __attribute__((ext_vector_type(8)))  int          v8i;
typedef __attribute__((ext_vector_type(4)))  unsigned int v4u;

#define N_NODES 4096
#define NWORDS  (N_NODES / 32)
#define CIN     256
#define COUT    256
#define HEADS   4
#define CHEAD   64
#define ALPHA   0.2f

// ---------------------------------------------------------------------------
// Kernel 1: h = X @ W^T + b  (4096x256, K=256) via v_wmma_f32_16x16x32_f16.
// Also writes h16T[n][i] = f16(h[i][n]) so the aggregation kernel's B tiles
// (K = node index j) are contiguous in memory.
// ---------------------------------------------------------------------------
__global__ __launch_bounds__(256) void gat_gemm_h(
    const float* __restrict__ X, const float* __restrict__ W,
    const float* __restrict__ bias, float* __restrict__ h,
    _Float16* __restrict__ h16T)
{
    const int lane  = threadIdx.x & 31;
    const int wv    = threadIdx.x >> 5;
    const int itile = blockIdx.x;                 // 0..255
    const int m     = lane & 15;
    const int row   = itile * 16 + m;
    const int hiA   = (lane & 16) ? 16 : 0;       // A-layout K sub-offset
    const int hiB   = (lane & 16) ? 16 : 0;       // B-layout K sub-offset

    v8f acc0 = {}; v8f acc1 = {};

    for (int k = 0; k < CIN; k += 32) {
        v16h a;
        const float* ap0 = X + row * CIN + k + hiA;
        const float* ap1 = ap0 + 8;
        #pragma unroll
        for (int e = 0; e < 8; ++e) {
            a[e]     = (_Float16)ap0[e];
            a[8 + e] = (_Float16)ap1[e];
        }
        #pragma unroll
        for (int t = 0; t < 2; ++t) {
            const int n = wv * 32 + t * 16 + m;
            const float* bp = W + n * CIN + k + hiB;
            v16h b;
            #pragma unroll
            for (int e = 0; e < 16; ++e) b[e] = (_Float16)bp[e];
            if (t == 0)
                acc0 = __builtin_amdgcn_wmma_f32_16x16x32_f16(
                           false, a, false, b, (short)0, acc0, false, false);
            else
                acc1 = __builtin_amdgcn_wmma_f32_16x16x32_f16(
                           false, a, false, b, (short)0, acc1, false, false);
        }
    }

    const int rbase = (lane & 16) ? 8 : 0;
    #pragma unroll
    for (int t = 0; t < 2; ++t) {
        const int n  = wv * 32 + t * 16 + m;
        const float bb = bias[n];
        v8f acc = t ? acc1 : acc0;
        #pragma unroll
        for (int r = 0; r < 8; ++r) {
            const int i = itile * 16 + rbase + r;
            const float v = acc[r] + bb;
            h[i * COUT + n] = v;
            h16T[(size_t)n * N_NODES + i] = (_Float16)v;
        }
    }
}

// ---------------------------------------------------------------------------
// Kernel 2: e_src[h][i] = h[i,h,:]·a_src[h],  e_dst[h][i] = h[i,h,:]·a_dst[h]
// ---------------------------------------------------------------------------
__global__ __launch_bounds__(256) void gat_edge_logits(
    const float* __restrict__ h, const float* __restrict__ a,
    float* __restrict__ e_src, float* __restrict__ e_dst)
{
    const int g  = blockIdx.x * 256 + threadIdx.x;
    const int i  = g >> 2;
    const int hd = g & 3;
    const float* hp = h + i * COUT + hd * CHEAD;
    const float* as = a + hd * 2 * CHEAD;
    const float* ad = as + CHEAD;
    float s = 0.f, d = 0.f;
    #pragma unroll 8
    for (int c = 0; c < CHEAD; ++c) {
        const float v = hp[c];
        s += v * as[c];
        d += v * ad[c];
    }
    e_src[hd * N_NODES + i] = s;
    e_dst[hd * N_NODES + i] = d;
}

// ---------------------------------------------------------------------------
// Kernel 3: pack the 0/1 int32 adjacency into a bitmask (64 MB -> 2 MB, which
// stays resident in the 192 MB L2). Wave32 ballot: lane L supplies bit L.
// ---------------------------------------------------------------------------
__global__ __launch_bounds__(256) void gat_pack_adj(
    const int* __restrict__ adj, unsigned int* __restrict__ abits)
{
    const int lane = threadIdx.x & 31;
    const int wid  = (blockIdx.x * 256 + threadIdx.x) >> 5;   // global wave id
    // 4096 waves x 128 words = 524288 words total
    #pragma unroll 4
    for (int t = 0; t < 128; ++t) {
        const size_t w = (size_t)wid * 128 + t;
        const int v = adj[w * 32 + lane];
        const unsigned long long b = __ballot(v != 0);
        if (lane == 0) abits[w] = (unsigned int)b;
    }
}

// ---------------------------------------------------------------------------
// Kernel 4: masked softmax + aggregation, single pass over the bitmask.
// e_dst (4 heads x 16 KB) is DMA'd once per workgroup into LDS by the Tensor
// Data Mover (tensor_load_to_lds + s_wait_tensorcnt), then served by
// ds_load_b128 in the hot loop.
// Shift m_i = leaky(e_src[i] + max_j e_dst[j]) upper-bounds the masked row
// max (leaky_relu is monotone) so p = exp(logit - m_i) <= 1 fits f16.
// ---------------------------------------------------------------------------
__global__ __launch_bounds__(256) void gat_attn(
    const unsigned int* __restrict__ abits, const float* __restrict__ e_src,
    const float* __restrict__ e_dst, const _Float16* __restrict__ h16T,
    float* __restrict__ out)
{
    __shared__ float lds_ed[HEADS * N_NODES];   // 64 KB, sole LDS alloc -> offset 0

    const int lane  = threadIdx.x & 31;
    const int wv    = threadIdx.x >> 5;
    const int head  = wv & 3;
    const int itile = blockIdx.x * 2 + (wv >> 2);
    const int m     = lane & 15;                 // local row (A-layout M)
    const int i     = itile * 16 + m;
    const int hiJ   = (lane & 16) ? 16 : 0;      // A-layout j sub-offset
    const int hiB   = (lane & 16) ? 16 : 0;      // B-layout j sub-offset

    // --- TDM: DMA e_dst[4][4096] f32 into LDS offset 0 (one wave issues) ---
    if (wv == 0) {
        const unsigned long long ga = (unsigned long long)(uintptr_t)e_dst;
        v4u g0;
        g0[0] = 1u;                                    // count=1, user mode
        g0[1] = 0u;                                    // lds_addr = 0
        g0[2] = (unsigned int)(ga & 0xFFFFFFFFu);      // global_addr[31:0]
        g0[3] = (unsigned int)((ga >> 32) & 0x1FFFFFFu) | (2u << 30); // addr[56:32], type=2
        v8i g1;
        g1[0] = (int)(2u << 16);                       // data_size = 4 bytes
        g1[1] = (int)(((unsigned)N_NODES & 0xFFFFu) << 16); // tensor_dim0 lo16
        g1[2] = (int)((unsigned)HEADS << 16);          // dim0 hi16=0 | tensor_dim1 lo16
        g1[3] = (int)(((unsigned)N_NODES & 0xFFFFu) << 16); // dim1 hi16=0 | tile_dim0
        g1[4] = (int)HEADS;                            // tile_dim1=4, tile_dim2=0
        g1[5] = (int)N_NODES;                          // tensor_dim0_stride[31:0]
        g1[6] = (int)(((unsigned)N_NODES & 0xFFFFu) << 16); // stride0 hi=0 | stride1 lo16
        g1[7] = 0;                                     // stride1 hi32
        v4i z4 = {};
        v8i z8 = {};
        __builtin_amdgcn_tensor_load_to_lds(g0, g1, z4, z4, z8, 0);
        __builtin_amdgcn_s_wait_tensorcnt(0);
    }
    __syncthreads();

    // Upper bound on e_dst for this head (LDS scan + wave32 reduction)
    float mx = -1e30f;
    for (int j = lane; j < N_NODES; j += 32)
        mx = fmaxf(mx, lds_ed[head * N_NODES + j]);
    #pragma unroll
    for (int s = 16; s; s >>= 1) mx = fmaxf(mx, __shfl_xor(mx, s, 32));

    const float es = e_src[head * N_NODES + i];
    const float xb = es + mx;
    const float mrow = xb > 0.f ? xb : ALPHA * xb;   // per-row softmax shift

    v8f acc[4] = {{}, {}, {}, {}};
    float lsum = 0.f;
    const unsigned int* arow = abits + (size_t)i * NWORDS;
    const _Float16*     hb   = h16T + (size_t)(head * CHEAD) * N_NODES;

    for (int jb = 0; jb < N_NODES; jb += 32) {
        const unsigned int word = arow[jb >> 5];     // 32 adjacency bits
        __builtin_prefetch(hb + (size_t)m * N_NODES + jb + 256, 0, 1);

        // Build probability tile directly in A-fragment layout
        v16h pa;
        #pragma unroll
        for (int g = 0; g < 2; ++g) {
            const int j0 = jb + g * 8 + hiJ;
            const v4f ev0 = *(const v4f*)&lds_ed[head * N_NODES + j0];
            const v4f ev1 = *(const v4f*)&lds_ed[head * N_NODES + j0 + 4];
            #pragma unroll
            for (int e = 0; e < 4; ++e) {
                float x = es + ev0[e];
                float lr = x > 0.f ? x : ALPHA * x;
                unsigned bit0 = (word >> (g * 8 + hiJ + e)) & 1u;
                _Float16 ph = (_Float16)(bit0 ? __expf(lr - mrow) : 0.f);
                pa[g * 8 + e] = ph;
                lsum += (float)ph;

                x = es + ev1[e];
                lr = x > 0.f ? x : ALPHA * x;
                unsigned bit1 = (word >> (g * 8 + hiJ + 4 + e)) & 1u;
                ph = (_Float16)(bit1 ? __expf(lr - mrow) : 0.f);
                pa[g * 8 + 4 + e] = ph;
                lsum += (float)ph;
            }
        }

        // acc[16 x 64] += P(16x32) @ H(32x64): 4 WMMAs over the column tiles
        #pragma unroll
        for (int ct = 0; ct < 4; ++ct) {
            const int c = ct * 16 + m;
            const v16h b = *(const v16h*)(hb + (size_t)c * N_NODES + jb + hiB);
            acc[ct] = __builtin_amdgcn_wmma_f32_16x16x32_f16(
                          false, pa, false, b, (short)0, acc[ct], false, false);
        }
    }

    // Row sums: lanes L and L+16 hold partials of the same row
    lsum += __shfl_xor(lsum, 16, 32);

    const int rbase = (lane & 16) ? 8 : 0;
    float inv[8];
    #pragma unroll
    for (int r = 0; r < 8; ++r) {
        const float l = __shfl(lsum, rbase + r, 32);
        inv[r] = l > 0.f ? 1.f / l : 0.f;
    }

    #pragma unroll
    for (int ct = 0; ct < 4; ++ct) {
        #pragma unroll
        for (int r = 0; r < 8; ++r) {
            const int row = itile * 16 + rbase + r;
            out[(size_t)row * COUT + head * CHEAD + ct * 16 + m] = acc[ct][r] * inv[r];
        }
    }
}

// ---------------------------------------------------------------------------
extern "C" void kernel_launch(void* const* d_in, const int* in_sizes, int n_in,
                              void* d_out, int out_size, void* d_ws, size_t ws_size,
                              hipStream_t stream) {
    const float* X   = (const float*)d_in[0];   // node_feats (4096,256) f32
    const int*   adj = (const int*)  d_in[1];   // adj (1,4096,4096) i32
    const float* W   = (const float*)d_in[2];   // W (256,256) f32
    const float* b   = (const float*)d_in[3];   // b (256,) f32
    const float* a   = (const float*)d_in[4];   // a (4,128) f32
    float* out = (float*)d_out;                 // (1,4096,256) f32

    char* ws = (char*)d_ws;
    float*        h     = (float*)ws;                                        // 4 MB
    _Float16*     h16T  = (_Float16*)(ws + (size_t)4096 * 1024);             // 2 MB
    float*        e_src = (float*)(ws + (size_t)6144 * 1024);                // 64 KB
    float*        e_dst = (float*)(ws + (size_t)6208 * 1024);                // 64 KB
    unsigned int* abits = (unsigned int*)(ws + (size_t)6272 * 1024);         // 2 MB

    gat_pack_adj   <<<512, 256, 0, stream>>>(adj, abits);
    gat_gemm_h     <<<N_NODES / 16, 256, 0, stream>>>(X, W, b, h, h16T);
    gat_edge_logits<<<(N_NODES * HEADS) / 256, 256, 0, stream>>>(h, a, e_src, e_dst);
    gat_attn       <<<N_NODES / 32, 256, 0, stream>>>(abits, e_src, e_dst, h16T, out);
}
